// SpectralConv2d_MHD_74406013435959
// MI455X (gfx1250) — matrix operand order repaired and checked by reading, not compiled, not run on hardware
//
#include <hip/hip_runtime.h>

typedef __attribute__((ext_vector_type(16))) _Float16 v16h;
typedef __attribute__((ext_vector_type(8)))  float    v8f;
typedef unsigned int u32;
typedef __attribute__((ext_vector_type(4)))  u32      u32x4;

#define DEV_INLINE __device__ __forceinline__

namespace {
constexpr int   Bn = 8, Mn = 256, Nn = 256, In = 64, On = 64, Kat = 4, MODES = 32;
constexpr float PI_F = 3.14159265358979323846f;
}

// ---------------------------------------------------------------------------
// WMMA fragment loaders (per CDNA5 ISA 16-bit A 16x32 / B 32x16 VGPR layouts).
// A: row-major, lane%16 = M row, lane/16 selects K interleave; pairs are
//    contiguous -> two b128 loads per fragment.
// B: supplied TRANSPOSED (Bt[col][k], i.e. column-major B) so lane%16 = N col
//    and the lane's 16 K-halves sit in two contiguous runs of 8 -> two b128 loads.
// ---------------------------------------------------------------------------
DEV_INLINE v16h load_a_frag(const _Float16* ap, int half) {
  union { v16h h; u32x4 q[2]; } a;
  a.q[0] = *(const u32x4*)(ap + half * 8);        // K = half*8 .. +7
  a.q[1] = *(const u32x4*)(ap + 16 + half * 8);   // K = 16+half*8 .. +7
  return a.h;
}
DEV_INLINE v16h load_b_frag(const _Float16* bp, int half) {
  union { v16h h; u32x4 q[2]; } b;
  b.q[0] = *(const u32x4*)(bp + half * 8);        // K = half*8 .. +7
  b.q[1] = *(const u32x4*)(bp + 16 + half * 8);   // K = 16+half*8 .. +7
  return b.h;
}

// ---------------------------------------------------------------------------
// Generic batched NT GEMM:  C[z][row][col] (+)= A[z][row][K] * Bt[z][col][K]
// One wave per 16x16 C tile, K in steps of 32 via v_wmma_f32_16x16x32_f16.
// C address: (z/divC)*sC1 + (z%divC)*sC2 + row*ldc + col  (col stride 1).
// ---------------------------------------------------------------------------
template <bool OUT_F16, bool ACCUM>
__global__ __launch_bounds__(128)
void k_wmma_gemm_nt(const _Float16* __restrict__ A, const _Float16* __restrict__ Bt,
                    void* __restrict__ Cv, int lda, int ldb, int ldc, int Kdim,
                    long strideA, long strideB, int divC, long sC1, long sC2) {
  const int lane = threadIdx.x & 31;
  const int r16  = lane & 15, half = lane >> 4;
  const int tileM = blockIdx.y * blockDim.y + threadIdx.y;
  const int tileN = blockIdx.x;
  const int z     = blockIdx.z;

  const _Float16* Ab = A  + z * strideA + (long)(tileM * 16 + r16) * lda;
  const _Float16* Bb = Bt + z * strideB + (long)(tileN * 16 + r16) * ldb;
  const long coff = (long)(z / divC) * sC1 + (long)(z % divC) * sC2
                  + (long)(tileM * 16 + 8 * half) * ldc + tileN * 16 + r16;

  float*    Cf = (float*)Cv;
  _Float16* Ch = (_Float16*)Cv;

  v8f c = {0.f, 0.f, 0.f, 0.f, 0.f, 0.f, 0.f, 0.f};
  if constexpr (ACCUM) {
#pragma unroll
    for (int v = 0; v < 8; ++v) c[v] = Cf[coff + (long)v * ldc];
  }

#pragma unroll 4
  for (int k0 = 0; k0 < Kdim; k0 += 32) {
    v16h a = load_a_frag(Ab + k0, half);
    v16h b = load_b_frag(Bb + k0, half);
    c = __builtin_amdgcn_wmma_f32_16x16x32_f16(false, a, false, b, (short)0, c,
                                               false, false);
  }

#pragma unroll
  for (int v = 0; v < 8; ++v) {
    long ad = coff + (long)v * ldc;           // C row = v + 8*half (ISA layout)
    if constexpr (OUT_F16) Ch[ad] = (_Float16)c[v];
    else                   Cf[ad] = c[v];
  }
}

// ---------------------------------------------------------------------------
// Pack: x (B,M,N,I) f32 -> xt16[b][i][m][n] and xtT16[b][i][n][m], both f16.
// ---------------------------------------------------------------------------
__global__ void k_pack_x(const float* __restrict__ x, _Float16* __restrict__ xt,
                         _Float16* __restrict__ xtT) {
  long idx = (long)blockIdx.x * 256 + threadIdx.x;
  if (idx >= (long)Bn * Mn * Nn * In) return;
  int i = idx % In; long t = idx / In;
  int n = t % Nn; t /= Nn;
  int m = t % Mn; int b = (int)(t / Mn);
  _Float16 h = (_Float16)x[idx];
  xt [(((long)b * In + i) * Mn + m) * Nn + n] = h;
  xtT[(((long)b * In + i) * Nn + n) * Mn + m] = h;
}

// ---------------------------------------------------------------------------
// Basis / static-weight prep. 6 regions of 16384 elements each.
//  FyT[c][n]: forward 256-pt rfft basis (c<32: cos, c>=32: -sin) * 1/16
//  GxT[c][m]: forward 510-pt odd-extension basis (analytic collapse) * 1/sqrt(510)
//  CyA[n][c]: inverse (irfft, 32 modes) basis * 1/16
//  HxA[m][c]: inverse 510-pt basis (first 256 rows) * 1/sqrt(510)
//  w1T[j][k] = w1[k][j] (f16),  w2T[o][k] = w2[k][o] (f16)
// ---------------------------------------------------------------------------
__global__ void k_prep_basis(const float* __restrict__ w1, const float* __restrict__ w2,
                             _Float16* FyT, _Float16* GxT, _Float16* CyA,
                             _Float16* HxA, _Float16* w1T, _Float16* w2T) {
  int idx = blockIdx.x * 256 + threadIdx.x;
  int region = idx >> 14;
  int r = idx & 16383;
  const float s16 = 0.0625f;          // 1/sqrt(256)
  const float nu  = 0.04428155940f;   // 1/sqrt(510)
  if (region == 0) {                  // FyT [64][256]
    int c = r >> 8, n = r & 255, y = c & 31;
    float ang = 2.f * PI_F * (float)(n * y) / 256.f;
    float v = (c < 32) ? cosf(ang) : -sinf(ang);
    FyT[r] = (_Float16)(v * s16);
  } else if (region == 1) {           // GxT [64][256]
    int c = r >> 8, m = r & 255, xm = c & 31;
    float v;
    if (c < 32) v = (m == 0) ? 1.f : ((m == 255) ? ((xm & 1) ? -1.f : 1.f) : 0.f);
    else        v = (m == 0 || m == 255) ? 0.f : -2.f * sinf(PI_F * (float)(m * xm) / 255.f);
    GxT[r] = (_Float16)(v * nu);
  } else if (region == 2) {           // CyA [256][64]
    int n = r >> 6, c = r & 63, y = c & 31;
    float ang = 2.f * PI_F * (float)(n * y) / 256.f;
    float v = (c < 32) ? ((y == 0) ? 1.f : 2.f * cosf(ang))
                       : ((y == 0) ? 0.f : -2.f * sinf(ang));
    CyA[r] = (_Float16)(v * s16);
  } else if (region == 3) {           // HxA [256][64]
    int m = r >> 6, c = r & 63, xm = c & 31;
    float ang = PI_F * (float)(m * xm) / 255.f;
    float v = (c < 32) ? ((xm == 0) ? 1.f : 2.f * cosf(ang))
                       : ((xm == 0) ? 0.f : -2.f * sinf(ang));
    HxA[r] = (_Float16)(v * nu);
  } else if (region == 4) {           // w1T [256][64]
    int j = r >> 6, k = r & 63;
    w1T[r] = (_Float16)w1[k * 256 + j];
  } else {                            // w2T [64][256]
    int o = r >> 8, k = r & 255;
    w2T[r] = (_Float16)w2[k * 64 + o];
  }
}

// ---------------------------------------------------------------------------
// Per-batch complex mixing weights packed for the real NT GEMM:
//  W2[b][mode][o'=p1*64+o][c=p0*64+i], p1: out re/im, p0: in re/im.
//  re/im = sum_k att[b,k] * fw[k,i,o,mode,{0,1}]
// ---------------------------------------------------------------------------
__global__ void k_wprep(const float* __restrict__ att, const float* __restrict__ fw,
                        _Float16* __restrict__ W2) {
  long idx = (long)blockIdx.x * 256 + threadIdx.x;
  if (idx >= (long)Bn * MODES * 128 * 128) return;
  int c = idx % 128; long t = idx / 128;
  int op = t % 128;  t /= 128;
  int y = t % MODES; int b = (int)(t / MODES);
  int p1 = op >> 6, o = op & 63;
  int p0 = c  >> 6, i = c  & 63;
  float re = 0.f, im = 0.f;
  for (int k = 0; k < Kat; ++k) {
    float a = att[b * Kat + k];
    const float* f = fw + ((((long)k * In + i) * On + o) * MODES + y) * 2;
    re += a * f[0];
    im += a * f[1];
  }
  float v = (p1 == 0) ? ((p0 == 0) ? re : -im) : ((p0 == 0) ? im : re);
  W2[idx] = (_Float16)v;
}

// T1[b][i][r][p*32+mode] -> T2[b][mode][r][p*64+i]
__global__ void k_repack_modes(const _Float16* __restrict__ T1, _Float16* __restrict__ T2) {
  long idx = (long)blockIdx.x * 256 + threadIdx.x;
  if (idx >= (long)Bn * In * 256 * 64) return;
  int c = idx % 64; long t = idx / 64;
  int r = t % 256;  t /= 256;
  int i = t % In;   int b = (int)(t / In);
  int p = c >> 5, mode = c & 31;
  T2[((long)(b * MODES + mode) * 256 + r) * 128 + p * 64 + i] = T1[idx];
}

// U[b][mode][r][p*64+o] -> U2[b][r][o][p*32+mode]
__global__ void k_repack_out(const _Float16* __restrict__ U, _Float16* __restrict__ U2) {
  long idx = (long)blockIdx.x * 256 + threadIdx.x;
  if (idx >= (long)Bn * MODES * 256 * 128) return;
  int cp = idx % 128; long t = idx / 128;
  int r = t % 256;    t /= 256;
  int mode = t % MODES; int b = (int)(t / MODES);
  int p = cp >> 6, o = cp & 63;
  U2[(((long)b * 256 + r) * 64 + o) * 64 + p * 32 + mode] = U[idx];
}

// ---------------------------------------------------------------------------
// Fused MLP + LayerNorm. 256 threads (8 waves), 128 points per block.
//  H1 = relu(H @ w1 + b1) staged in LDS (f16, 128x256 = 64KB)
//  H2 = H1 @ w2 + b2 staged in LDS (f32, 128x64 = 32KB)
//  LN epilogue per point -> d_out.
// ---------------------------------------------------------------------------
__global__ __launch_bounds__(256)
void k_mlp_ln(const float* __restrict__ Hacc,
              const _Float16* __restrict__ w1T, const float* __restrict__ b1,
              const _Float16* __restrict__ w2T, const float* __restrict__ b2,
              const float* __restrict__ gamma, const float* __restrict__ beta,
              float* __restrict__ out) {
  extern __shared__ char smem[];
  _Float16* H1 = (_Float16*)smem;                       // [128][256]
  float*    H2 = (float*)(smem + 128 * 256 * 2);        // [128][64]

  const int tid  = threadIdx.x;
  const int lane = tid & 31, w = tid >> 5;
  const int r16  = lane & 15, half = lane >> 4;
  const long p0  = (long)blockIdx.x * 128;

  // Load this wave's 16 input rows as two A fragments (K=0..31, 32..63), f32->f16.
  v16h aA, aB;
  const float* hp = Hacc + (p0 + w * 16 + r16) * 64;
#pragma unroll
  for (int j = 0; j < 8; ++j) {
    int kk = (j < 4) ? (half * 8 + 2 * j) : (16 + half * 8 + 2 * (j - 4));
    aA[2 * j]     = (_Float16)hp[kk];
    aA[2 * j + 1] = (_Float16)hp[kk + 1];
    aB[2 * j]     = (_Float16)hp[32 + kk];
    aB[2 * j + 1] = (_Float16)hp[32 + kk + 1];
  }

  // H1 = relu(H @ w1 + b1): 16 column tiles of 16, K=64.
  for (int nt = 0; nt < 16; ++nt) {
    v8f c = {0.f, 0.f, 0.f, 0.f, 0.f, 0.f, 0.f, 0.f};
    const _Float16* bp = w1T + (nt * 16 + r16) * 64;
    c = __builtin_amdgcn_wmma_f32_16x16x32_f16(false, aA, false, load_b_frag(bp, half),
                                               (short)0, c, false, false);
    c = __builtin_amdgcn_wmma_f32_16x16x32_f16(false, aB, false, load_b_frag(bp + 32, half),
                                               (short)0, c, false, false);
    int col = nt * 16 + r16;
    float bias = b1[col];
#pragma unroll
    for (int v = 0; v < 8; ++v) {
      float hv = fmaxf(c[v] + bias, 0.f);
      H1[(w * 16 + v + 8 * half) * 256 + col] = (_Float16)hv;
    }
  }
  __syncthreads();

  // H2 = H1 @ w2 + b2: 4 column tiles of 16, K=256 (A fragments from LDS).
  for (int ot = 0; ot < 4; ++ot) {
    v8f c = {0.f, 0.f, 0.f, 0.f, 0.f, 0.f, 0.f, 0.f};
#pragma unroll
    for (int ks = 0; ks < 8; ++ks) {
      const _Float16* ap = &H1[(w * 16 + r16) * 256 + ks * 32];
      const _Float16* bp = w2T + (ot * 16 + r16) * 256 + ks * 32;
      c = __builtin_amdgcn_wmma_f32_16x16x32_f16(false, load_a_frag(ap, half),
                                                 false, load_b_frag(bp, half),
                                                 (short)0, c, false, false);
    }
    int col = ot * 16 + r16;
    float bias = b2[col];
#pragma unroll
    for (int v = 0; v < 8; ++v)
      H2[(w * 16 + v + 8 * half) * 64 + col] = c[v] + bias;
  }
  __syncthreads();

  // LayerNorm per point.
  if (tid < 128) {
    const float* hr = &H2[tid * 64];
    float mu = 0.f;
    for (int o = 0; o < 64; ++o) mu += hr[o];
    mu *= (1.f / 64.f);
    float var = 0.f;
    for (int o = 0; o < 64; ++o) { float d = hr[o] - mu; var += d * d; }
    var *= (1.f / 64.f);
    float rs = rsqrtf(var + 1e-5f);
    float* op = out + (p0 + tid) * 64;
    for (int o = 0; o < 64; ++o) op[o] = (hr[o] - mu) * rs * gamma[o] + beta[o];
  }
}

// ---------------------------------------------------------------------------
extern "C" void kernel_launch(void* const* d_in, const int* in_sizes, int n_in,
                              void* d_out, int out_size, void* d_ws, size_t ws_size,
                              hipStream_t stream) {
  (void)in_sizes; (void)n_in; (void)out_size; (void)ws_size;
  const float* x     = (const float*)d_in[0];
  const float* att   = (const float*)d_in[1];
  const float* fw_y  = (const float*)d_in[2];
  const float* fw_x  = (const float*)d_in[3];
  const float* w1    = (const float*)d_in[4];
  const float* b1    = (const float*)d_in[5];
  const float* w2    = (const float*)d_in[6];
  const float* b2    = (const float*)d_in[7];
  const float* gamma = (const float*)d_in[8];
  const float* beta  = (const float*)d_in[9];
  float* out = (float*)d_out;

  size_t off = 0;
  auto carve = [&](size_t bytes) -> char* {
    char* p = (char*)d_ws + off;
    off += (bytes + 255) & ~(size_t)255;
    return p;
  };
  _Float16* xt16  = (_Float16*)carve((size_t)Bn * In * Mn * Nn * 2);   // 67MB
  _Float16* xtT16 = (_Float16*)carve((size_t)Bn * In * Nn * Mn * 2);   // 67MB
  _Float16* T1    = (_Float16*)carve((size_t)Bn * In * 256 * 64 * 2);  // fwd-DFT out
  _Float16* T2    = (_Float16*)carve((size_t)Bn * MODES * 256 * 128 * 2);
  _Float16* Wb    = (_Float16*)carve((size_t)Bn * MODES * 128 * 128 * 2);
  _Float16* Ob    = (_Float16*)carve((size_t)Bn * MODES * 256 * 128 * 2);
  _Float16* O2    = (_Float16*)carve((size_t)Bn * 256 * 64 * 64 * 2);
  float*    Hacc  = (float*)   carve((size_t)Bn * Mn * Nn * On * 4);   // 134MB
  _Float16* FyT   = (_Float16*)carve(64 * 256 * 2);
  _Float16* GxT   = (_Float16*)carve(64 * 256 * 2);
  _Float16* CyA   = (_Float16*)carve(256 * 64 * 2);
  _Float16* HxA   = (_Float16*)carve(256 * 64 * 2);
  _Float16* w1T   = (_Float16*)carve(256 * 64 * 2);
  _Float16* w2T   = (_Float16*)carve(64 * 256 * 2);

  const dim3 gb(32, 4, 1);
  const long MNO = (long)Mn * Nn * On;

  // Prep
  k_pack_x<<<131072, 256, 0, stream>>>(x, xt16, xtT16);
  k_prep_basis<<<384, 256, 0, stream>>>(w1, w2, FyT, GxT, CyA, HxA, w1T, w2T);

  // ---- Y branch (rfft along N) ----
  k_wprep<<<16384, 256, 0, stream>>>(att, fw_y, Wb);
  // A1: per (b,i): [m=256][n=256] x FyT -> T1[b][i][m][64]
  k_wmma_gemm_nt<true, false><<<dim3(4, 4, 512), gb, 0, stream>>>(
      xt16, FyT, T1, 256, 256, 64, 256, 65536L, 0L, 1, 16384L, 0L);
  k_repack_modes<<<32768, 256, 0, stream>>>(T1, T2);
  // A2: per (b,y): [m=256][128] x Wy -> Ob[b][y][m][128]
  k_wmma_gemm_nt<true, false><<<dim3(8, 4, 256), gb, 0, stream>>>(
      T2, Wb, Ob, 128, 128, 128, 128, 32768L, 16384L, 1, 32768L, 0L);
  k_repack_out<<<32768, 256, 0, stream>>>(Ob, O2);
  // A3 (inverse): per (b,m): CyA[n][64] x O2[b][m][o][64] -> Hacc[b][m][n][o]
  k_wmma_gemm_nt<false, false><<<dim3(4, 4, 2048), gb, 0, stream>>>(
      CyA, O2, Hacc, 64, 64, 64, 64, 0L, 4096L, 256, MNO, (long)Nn * On);

  // ---- X branch (odd-extended rfft along M) ----
  k_wprep<<<16384, 256, 0, stream>>>(att, fw_x, Wb);
  // B1: per (b,i): [n=256][m=256] x GxT -> T1[b][i][n][64]
  k_wmma_gemm_nt<true, false><<<dim3(4, 4, 512), gb, 0, stream>>>(
      xtT16, GxT, T1, 256, 256, 64, 256, 65536L, 0L, 1, 16384L, 0L);
  k_repack_modes<<<32768, 256, 0, stream>>>(T1, T2);
  k_wmma_gemm_nt<true, false><<<dim3(8, 4, 256), gb, 0, stream>>>(
      T2, Wb, Ob, 128, 128, 128, 128, 32768L, 16384L, 1, 32768L, 0L);
  k_repack_out<<<32768, 256, 0, stream>>>(Ob, O2);
  // B3 (inverse, ACCUMULATE): per (b,n): HxA[m][64] x O2[b][n][o][64] += Hacc[b][m][n][o]
  k_wmma_gemm_nt<false, true><<<dim3(4, 4, 2048), gb, 0, stream>>>(
      HxA, O2, Hacc, 64, 64, Nn * On, 64, 0L, 4096L, 256, MNO, (long)On);

  // ---- MLP + LayerNorm ----
  k_mlp_ln<<<4096, 256, 128 * 256 * 2 + 128 * 64 * 4, stream>>>(
      Hacc, w1T, b1, w2T, b2, gamma, beta, out);
}
